// MultiHeadAttention_50732153700976
// MI455X (gfx1250) — matrix operand order repaired and checked by reading, not compiled, not run on hardware
//
#include <hip/hip_runtime.h>
#include <math.h>

typedef _Float16 half_t;
typedef __attribute__((ext_vector_type(16))) _Float16 v16h;
typedef __attribute__((ext_vector_type(8)))  _Float16 v8h;
typedef __attribute__((ext_vector_type(4)))  _Float16 v4h;
typedef __attribute__((ext_vector_type(8)))  float    v8f;
typedef __attribute__((ext_vector_type(2)))  float    vf2;
typedef __attribute__((ext_vector_type(4)))  float    vf4;

#define B_  4
#define N_  1024
#define D_  1024
#define H_  16
#define HD_ 64
#define SCALE_ 0.125f   // 1/sqrt(64)

// workspace layout (in half elements)
#define OFF_XH     ((size_t)0)
#define OFF_WQKVT  (OFF_XH     + (size_t)B_*N_*D_)
#define OFF_WPROJT (OFF_WQKVT  + (size_t)3*D_*D_)
#define OFF_Q      (OFF_WPROJT + (size_t)D_*D_)
#define OFF_K      (OFF_Q      + (size_t)B_*H_*N_*HD_)
#define OFF_VT     (OFF_K      + (size_t)B_*H_*N_*HD_)
#define OFF_AO     (OFF_VT     + (size_t)B_*H_*HD_*N_)
// total = 25,165,824 halves = 48 MB

#define WMMA_F16(a, b, c) \
    __builtin_amdgcn_wmma_f32_16x16x32_f16(false, (a), false, (b), (short)0, (c), false, false)

// ---------------- helpers: WMMA fragment loads ----------------

// A fragment (16x32 f16, row-major source, per-lane row aRow):
// element i -> k = kbase + (i<8 ? i : i+8) + 8*half  => two 16B loads
__device__ __forceinline__ v16h load_a_h(const half_t* __restrict__ aRow, int kbase, int hf) {
    v8h lo = *(const v8h*)(aRow + kbase + hf * 8);
    v8h hi = *(const v8h*)(aRow + kbase + 16 + hf * 8);
    return __builtin_shufflevector(lo, hi, 0,1,2,3,4,5,6,7,8,9,10,11,12,13,14,15);
}

// A fragment from f32 source (attn probabilities), non-temporal, converted in-register
__device__ __forceinline__ v16h load_a_f32_nt(const float* __restrict__ aRow, int kbase, int hf) {
    v16h a;
#pragma unroll
    for (int i = 0; i < 8; i += 2) {
        vf2 lo = __builtin_nontemporal_load((const vf2*)(aRow + kbase + hf * 8 + i));
        vf2 hi = __builtin_nontemporal_load((const vf2*)(aRow + kbase + 16 + hf * 8 + i));
        a[i]     = (half_t)lo[0];
        a[i + 1] = (half_t)lo[1];
        a[i + 8] = (half_t)hi[0];
        a[i + 9] = (half_t)hi[1];
    }
    return a;
}

// B fragment (32x16, K-contiguous source): element i -> k = kbase + (i<8 ? i : i+8) + 8*half, two 16B loads
__device__ __forceinline__ v16h load_b_h(const half_t* __restrict__ bRow, int kbase, int hf) {
    v8h lo = *(const v8h*)(bRow + kbase + hf * 8);
    v8h hi = *(const v8h*)(bRow + kbase + 16 + hf * 8);
    return __builtin_shufflevector(lo, hi, 0,1,2,3,4,5,6,7,8,9,10,11,12,13,14,15);
}

// ---------------- conversion kernels ----------------

__global__ void cvt_f32_f16_x4(const float* __restrict__ src, half_t* __restrict__ dst) {
    size_t i = ((size_t)blockIdx.x * blockDim.x + threadIdx.x) * 4;
    vf4 v = *(const vf4*)(src + i);
    v4h h;
    h[0] = (half_t)v[0]; h[1] = (half_t)v[1]; h[2] = (half_t)v[2]; h[3] = (half_t)v[3];
    *(v4h*)(dst + i) = h;
}

// src: K x Nc f32 row-major  ->  dst: Nc x K f16 row-major (transposed, coalesced both sides)
__global__ void transpose_cvt(const float* __restrict__ src, half_t* __restrict__ dst,
                              int K, int Nc) {
    __shared__ float tile[32][33];
    int n0 = blockIdx.x * 32, k0 = blockIdx.y * 32;
    int tx = threadIdx.x, ty = threadIdx.y;          // 32 x 8
#pragma unroll
    for (int r = 0; r < 32; r += 8)
        tile[ty + r][tx] = src[(size_t)(k0 + ty + r) * Nc + n0 + tx];
    __syncthreads();
#pragma unroll
    for (int r = 0; r < 32; r += 8)
        dst[(size_t)(n0 + ty + r) * K + k0 + tx] = (half_t)tile[tx][ty + r];
}

// ---------------- GEMM 1: qkv = x @ w_qkv + b, routed into q / k / vT (f16) ----------------
// 2x2 register blocking: each wave computes a 32x32 output tile (4 WMMAs per k-step).

__global__ void gemm_qkv(const half_t* __restrict__ xh, const half_t* __restrict__ wT,
                         const float* __restrict__ bias,
                         half_t* __restrict__ qbuf, half_t* __restrict__ kbuf,
                         half_t* __restrict__ vtb) {
    int wid  = (int)((blockIdx.x * blockDim.x + threadIdx.x) >> 5);
    int lane = threadIdx.x & 31;
    int hf = lane >> 4, l15 = lane & 15;
    int mt2 = wid & 127;          // 128 tiles of 32 in M (4096)
    int nt2 = wid >> 7;           // 96 tiles of 32 in N (3072)
    int mrow = mt2 * 32, ncol = nt2 * 32;

    const half_t* aRow[2] = { xh + (size_t)(mrow + l15) * D_,
                              xh + (size_t)(mrow + 16 + l15) * D_ };
    const half_t* bRow[2] = { wT + (size_t)(ncol + l15) * D_,
                              wT + (size_t)(ncol + 16 + l15) * D_ };

    v8f c[2][2] = {};
    for (int k0 = 0; k0 < D_; k0 += 32) {
        v16h a0 = load_a_h(aRow[0], k0, hf);
        v16h a1 = load_a_h(aRow[1], k0, hf);
        v16h b0 = load_b_h(bRow[0], k0, hf);
        v16h b1 = load_b_h(bRow[1], k0, hf);
        c[0][0] = WMMA_F16(a0, b0, c[0][0]);
        c[0][1] = WMMA_F16(a0, b1, c[0][1]);
        c[1][0] = WMMA_F16(a1, b0, c[1][0]);
        c[1][1] = WMMA_F16(a1, b1, c[1][1]);
    }

#pragma unroll
    for (int ni = 0; ni < 2; ++ni) {
        int nglob = ncol + ni * 16 + l15;
        float bv  = bias[nglob];
        int which = nglob >> 10;        // 0=q 1=k 2=v
        int rem   = nglob & 1023;
        int h = rem >> 6, dd = rem & 63;
#pragma unroll
        for (int mi = 0; mi < 2; ++mi) {
#pragma unroll
            for (int r = 0; r < 8; ++r) {
                int mglob = mrow + mi * 16 + r + 8 * hf;
                int bb = mglob >> 10, nn = mglob & 1023;
                float v = c[mi][ni][r] + bv;
                if (which == 0)
                    qbuf[(((size_t)(bb * H_ + h) * N_) + nn) * HD_ + dd] = (half_t)v;
                else if (which == 1)
                    kbuf[(((size_t)(bb * H_ + h) * N_) + nn) * HD_ + dd] = (half_t)v;
                else
                    vtb[((size_t)(bb * H_ + h) * HD_ + dd) * N_ + nn] = (half_t)v;
            }
        }
    }
}

// ---------------- GEMM 2: S = q @ k^T * scale  (f32, non-temporal -> d_out attn) ----------------
// 2x2 blocking, K=64 (2 k-steps). attn stream (268 MB > 192 MB L2) bypasses cache residency.

__global__ void gemm_scores(const half_t* __restrict__ qbuf, const half_t* __restrict__ kbuf,
                            float* __restrict__ attn) {
    int wid  = (int)((blockIdx.x * blockDim.x + threadIdx.x) >> 5);
    int lane = threadIdx.x & 31;
    int hf = lane >> 4, l15 = lane & 15;
    int bh = wid >> 10;             // 64 (b,h) pairs
    int t  = wid & 1023;            // 32x32 tiles of 32 per (b,h)
    int mt2 = t & 31, nt2 = t >> 5;
    int mrow = mt2 * 32, ncol = nt2 * 32;

    const half_t* aRow[2] = { qbuf + ((size_t)bh * N_ + mrow + l15) * HD_,
                              qbuf + ((size_t)bh * N_ + mrow + 16 + l15) * HD_ };
    const half_t* bRow[2] = { kbuf + ((size_t)bh * N_ + ncol + l15) * HD_,
                              kbuf + ((size_t)bh * N_ + ncol + 16 + l15) * HD_ };

    v8f c[2][2] = {};
#pragma unroll
    for (int k0 = 0; k0 < HD_; k0 += 32) {
        v16h a0 = load_a_h(aRow[0], k0, hf);
        v16h a1 = load_a_h(aRow[1], k0, hf);
        v16h b0 = load_b_h(bRow[0], k0, hf);
        v16h b1 = load_b_h(bRow[1], k0, hf);
        c[0][0] = WMMA_F16(a0, b0, c[0][0]);
        c[0][1] = WMMA_F16(a0, b1, c[0][1]);
        c[1][0] = WMMA_F16(a1, b0, c[1][0]);
        c[1][1] = WMMA_F16(a1, b1, c[1][1]);
    }

    float* out = attn + (size_t)bh * N_ * N_;
#pragma unroll
    for (int mi = 0; mi < 2; ++mi) {
#pragma unroll
        for (int r = 0; r < 8; ++r) {
            size_t rowoff = (size_t)(mrow + mi * 16 + r + 8 * hf) * N_;
#pragma unroll
            for (int ni = 0; ni < 2; ++ni)
                __builtin_nontemporal_store(c[mi][ni][r] * SCALE_,
                                            out + rowoff + ncol + ni * 16 + l15);
        }
    }
}

// ---------------- softmax: one 256-thread block per row of 1024, in place (NT stream) --------

__global__ void softmax_rows(float* __restrict__ attn) {
    __shared__ float red[256];
    float* p = attn + (size_t)blockIdx.x * N_;
    int t = threadIdx.x;
    vf4 v = __builtin_nontemporal_load((const vf4*)(p + t * 4));

    float mx = fmaxf(fmaxf(v[0], v[1]), fmaxf(v[2], v[3]));
    red[t] = mx;
    __syncthreads();
#pragma unroll
    for (int s = 128; s > 0; s >>= 1) {
        if (t < s) red[t] = fmaxf(red[t], red[t + s]);
        __syncthreads();
    }
    mx = red[0];
    __syncthreads();

    vf4 e;
    e[0] = __expf(v[0] - mx); e[1] = __expf(v[1] - mx);
    e[2] = __expf(v[2] - mx); e[3] = __expf(v[3] - mx);
    red[t] = e[0] + e[1] + e[2] + e[3];
    __syncthreads();
#pragma unroll
    for (int s = 128; s > 0; s >>= 1) {
        if (t < s) red[t] += red[t + s];
        __syncthreads();
    }
    float inv = 1.0f / red[0];
    e[0] *= inv; e[1] *= inv; e[2] *= inv; e[3] *= inv;
    __builtin_nontemporal_store(e, (vf4*)(p + t * 4));
}

// ---------------- GEMM 3: out_head = attn @ v ----------------
// 2x4 blocking: each wave computes 32 rows x full head dim (64), so every fp32 attn
// element is loaded from memory exactly once (non-temporal).

__global__ void gemm_av(const float* __restrict__ attn, const half_t* __restrict__ vtb,
                        half_t* __restrict__ ao) {
    int wid  = (int)((blockIdx.x * blockDim.x + threadIdx.x) >> 5);
    int lane = threadIdx.x & 31;
    int hf = lane >> 4, l15 = lane & 15;
    int bh  = wid >> 5;             // 64 (b,h)
    int mt2 = wid & 31;             // 32 tiles of 32 rows
    int mrow = mt2 * 32;

    const float* aRow[2] = { attn + ((size_t)bh * N_ + mrow + l15) * N_,
                             attn + ((size_t)bh * N_ + mrow + 16 + l15) * N_ };
    const half_t* bBase = vtb + (size_t)bh * HD_ * N_;
    const half_t* bRow[4] = { bBase + (size_t)(l15) * N_,
                              bBase + (size_t)(16 + l15) * N_,
                              bBase + (size_t)(32 + l15) * N_,
                              bBase + (size_t)(48 + l15) * N_ };

    v8f c[2][4] = {};
    for (int k0 = 0; k0 < N_; k0 += 32) {
        v16h a0 = load_a_f32_nt(aRow[0], k0, hf);
        v16h a1 = load_a_f32_nt(aRow[1], k0, hf);
#pragma unroll
        for (int ni = 0; ni < 4; ++ni) {
            v16h b = load_b_h(bRow[ni], k0, hf);
            c[0][ni] = WMMA_F16(a0, b, c[0][ni]);
            c[1][ni] = WMMA_F16(a1, b, c[1][ni]);
        }
    }

    int b_ = bh >> 4, h = bh & 15;
#pragma unroll
    for (int mi = 0; mi < 2; ++mi) {
#pragma unroll
        for (int r = 0; r < 8; ++r) {
            int m = mrow + mi * 16 + r + 8 * hf;
            half_t* dst = ao + ((size_t)b_ * N_ + m) * D_ + h * HD_ + l15;
#pragma unroll
            for (int ni = 0; ni < 4; ++ni)
                dst[ni * 16] = (half_t)c[mi][ni][r];
        }
    }
}

// ---------------- GEMM 4: out = ao @ w_proj + b_proj  (f32 -> d_out) ----------------
// 2x2 register blocking.

__global__ void gemm_proj(const half_t* __restrict__ ao, const half_t* __restrict__ wT,
                          const float* __restrict__ bias, float* __restrict__ out) {
    int wid  = (int)((blockIdx.x * blockDim.x + threadIdx.x) >> 5);
    int lane = threadIdx.x & 31;
    int hf = lane >> 4, l15 = lane & 15;
    int mt2 = wid & 127;  // 128 tiles of 32 (M = 4096)
    int nt2 = wid >> 7;   // 32 tiles of 32 (N = 1024)
    int mrow = mt2 * 32, ncol = nt2 * 32;

    const half_t* aRow[2] = { ao + (size_t)(mrow + l15) * D_,
                              ao + (size_t)(mrow + 16 + l15) * D_ };
    const half_t* bRow[2] = { wT + (size_t)(ncol + l15) * D_,
                              wT + (size_t)(ncol + 16 + l15) * D_ };

    v8f c[2][2] = {};
    for (int k0 = 0; k0 < D_; k0 += 32) {
        v16h a0 = load_a_h(aRow[0], k0, hf);
        v16h a1 = load_a_h(aRow[1], k0, hf);
        v16h b0 = load_b_h(bRow[0], k0, hf);
        v16h b1 = load_b_h(bRow[1], k0, hf);
        c[0][0] = WMMA_F16(a0, b0, c[0][0]);
        c[0][1] = WMMA_F16(a0, b1, c[0][1]);
        c[1][0] = WMMA_F16(a1, b0, c[1][0]);
        c[1][1] = WMMA_F16(a1, b1, c[1][1]);
    }

#pragma unroll
    for (int ni = 0; ni < 2; ++ni) {
        int nglob = ncol + ni * 16 + l15;
        float bv  = bias[nglob];
#pragma unroll
        for (int mi = 0; mi < 2; ++mi)
#pragma unroll
            for (int r = 0; r < 8; ++r)
                out[(size_t)(mrow + mi * 16 + r + 8 * hf) * D_ + nglob] = c[mi][ni][r] + bv;
    }
}

// ---------------- host launcher ----------------

extern "C" void kernel_launch(void* const* d_in, const int* in_sizes, int n_in,
                              void* d_out, int out_size, void* d_ws, size_t ws_size,
                              hipStream_t stream) {
    const float* x      = (const float*)d_in[0];
    const float* w_qkv  = (const float*)d_in[1];
    const float* b_qkv  = (const float*)d_in[2];
    const float* w_proj = (const float*)d_in[3];
    const float* b_proj = (const float*)d_in[4];

    float* out  = (float*)d_out;                       // (B,N,D) f32
    float* attn = out + (size_t)B_ * N_ * D_;          // (B,H,N,N) f32

    half_t* ws     = (half_t*)d_ws;                    // needs 48 MB
    half_t* xh     = ws + OFF_XH;
    half_t* wqkvT  = ws + OFF_WQKVT;
    half_t* wprojT = ws + OFF_WPROJT;
    half_t* qbuf   = ws + OFF_Q;
    half_t* kbuf   = ws + OFF_K;
    half_t* vtb    = ws + OFF_VT;
    half_t* ao     = ws + OFF_AO;

    // 1) convert x to f16
    cvt_f32_f16_x4<<<(B_ * N_ * D_) / (256 * 4), 256, 0, stream>>>(x, xh);

    // 2) transpose+convert weights: (K x Nc) f32 -> (Nc x K) f16
    transpose_cvt<<<dim3(3 * D_ / 32, D_ / 32), dim3(32, 8), 0, stream>>>(w_qkv, wqkvT, D_, 3 * D_);
    transpose_cvt<<<dim3(D_ / 32, D_ / 32),     dim3(32, 8), 0, stream>>>(w_proj, wprojT, D_, D_);

    // 3) QKV GEMM: 128 x 96 wave tiles (32x32 each) = 12288 waves -> 1536 blocks
    gemm_qkv<<<1536, 256, 0, stream>>>(xh, wqkvT, b_qkv, qbuf, kbuf, vtb);

    // 4) scores: 64 bh * 1024 wave tiles = 65536 waves -> 8192 blocks
    gemm_scores<<<8192, 256, 0, stream>>>(qbuf, kbuf, attn);

    // 5) softmax over 65536 rows
    softmax_rows<<<B_ * H_ * N_, 256, 0, stream>>>(attn);

    // 6) attn @ v: 64 bh * 32 wave tiles (32x64 each) = 2048 waves -> 256 blocks
    gemm_av<<<256, 256, 0, stream>>>(attn, vtb, ao);

    // 7) projection: 128 x 32 wave tiles = 4096 waves -> 512 blocks
    gemm_proj<<<512, 256, 0, stream>>>(ao, wprojT, b_proj, out);
}
